// AttLayer_6193342841210
// MI455X (gfx1250) — matrix operand order repaired and checked by reading, not compiled, not run on hardware
//
#include <hip/hip_runtime.h>
#include <hip/hip_bf16.h>

// Problem constants (from reference): B=64, T=2048, D=512, A=128
#define B_N   64
#define T_N   2048
#define D_N   512
#define A_N   128
#define T_TILE 128                 // T rows per workgroup
#define N_CHUNK (T_N / T_TILE)     // 16 chunks per batch
#define WPITCH 520                 // bf16 elems per Wt row in LDS (512 + 8 pad -> no bank conflicts)
#define EPS_ 1e-7f

typedef __attribute__((ext_vector_type(16))) __bf16 v16bf;
typedef __attribute__((ext_vector_type(8)))  __bf16 v8bf;
typedef __attribute__((ext_vector_type(8)))  float  v8f;
typedef __attribute__((ext_vector_type(4)))  float  v4f;
typedef __attribute__((ext_vector_type(4)))  int    i32x4;

typedef __attribute__((address_space(1))) i32x4* as1_i32x4p;
typedef __attribute__((address_space(3))) i32x4* as3_i32x4p;

#if __has_builtin(__builtin_amdgcn_global_load_async_to_lds_b128)
#define HAVE_ASYNC 1
#else
#define HAVE_ASYNC 0
#endif

__device__ __forceinline__ void async_copy16(const void* g, void* l) {
#if HAVE_ASYNC
  __builtin_amdgcn_global_load_async_to_lds_b128(
      (as1_i32x4p)g, (as3_i32x4p)l, /*offset=*/0, /*cpol=*/0);
#else
  *(v8bf*)l = *(const v8bf*)g;   // fallback: plain 16B copy through VGPRs
#endif
}

__device__ __forceinline__ void wait_async0() {
#if HAVE_ASYNC
#if __has_builtin(__builtin_amdgcn_s_wait_asynccnt)
  __builtin_amdgcn_s_wait_asynccnt(0);
#else
  asm volatile("s_wait_asynccnt 0" ::: "memory");
#endif
#endif
}

// ---------------------------------------------------------------------------
// Kernel 0: zero accumulators; build Wt = bf16 transpose of W  (Wt[a][d])
// grid: (A_N*D_N)/256 blocks of 256
// ---------------------------------------------------------------------------
__global__ void attn_prep(const float* __restrict__ W, float* __restrict__ acc_zero,
                          __bf16* __restrict__ wt) {
  const int idx = blockIdx.x * blockDim.x + threadIdx.x;   // 0 .. 65535
  if (idx < B_N * D_N + B_N) acc_zero[idx] = 0.0f;         // numer[B*D] + denom[B]
  const int col = idx >> 9;        // a-index  0..127
  const int k   = idx & (D_N - 1); // d-index  0..511
  wt[col * D_N + k] = (__bf16)W[k * A_N + col];
}

// ---------------------------------------------------------------------------
// Kernel 1: fused  e = exp(tanh(x@W+b)@u);  numer += e*x;  denom += e
// grid: B_N*N_CHUNK blocks, 256 threads (8 waves), dynamic LDS
// ---------------------------------------------------------------------------
__global__ void __launch_bounds__(256)
attn_main(const float* __restrict__ x, const unsigned char* __restrict__ mask,
          const float* __restrict__ bp, const float* __restrict__ up,
          const __bf16* __restrict__ wtg,
          float* __restrict__ numerAcc, float* __restrict__ denomAcc) {
  extern __shared__ char smem[];
  __bf16* ldsW      = (__bf16*)smem;                          // [A_N][WPITCH] bf16
  float*  numer_lds = (float*)(smem + (size_t)A_N * WPITCH * 2); // [512]
  float*  e_lds     = numer_lds + D_N;                        // [T_TILE]
  float*  denom_lds = e_lds + T_TILE;                         // [1]

  const int tid   = threadIdx.x;
  const int bIdx  = blockIdx.x >> 4;        // batch
  const int chunk = blockIdx.x & (N_CHUNK - 1);
  const int wave  = tid >> 5;
  const int lane  = tid & 31;
  const int lm    = lane & 15;
  const int hh    = lane >> 4;              // half-wave select
  const int t0    = chunk * T_TILE + wave * 16;

  numer_lds[tid] = 0.0f;
  numer_lds[tid + 256] = 0.0f;
  if (tid == 0) *denom_lds = 0.0f;

  // ---- stage Wt into LDS with async copies (8192 x 16B chunks, 32 per thread),
  //      inserting the WPITCH pad via per-lane LDS destination addresses.
#pragma unroll 4
  for (int it = 0; it < 32; ++it) {
    const int c   = it * 256 + tid;     // chunk id
    const int col = c >> 6;             // 0..127
    const int kc  = (c & 63) << 3;      // bf16 k offset, multiple of 8 (16B)
    async_copy16(wtg + col * D_N + kc, ldsW + col * WPITCH + kc);
  }
  wait_async0();
  __syncthreads();

  // ---- GEMM: 16 rows (this wave) x A=128, K=512, bf16 WMMA 16x16x32
  // A-frag (16-bit 16x32): lanes 0-15 row M=lm hold K {0..7,16..23}; lanes 16-31 hold {8..15,24..31}
  const float* aBase = x + ((size_t)bIdx * T_N + t0 + lm) * D_N + hh * 8;
  v8f acc[8] = {};
  for (int k = 0; k < D_N; k += 32) {
    v4f a0 = *(const v4f*)(aBase + k);
    v4f a1 = *(const v4f*)(aBase + k + 4);
    v4f a2 = *(const v4f*)(aBase + k + 16);
    v4f a3 = *(const v4f*)(aBase + k + 20);
    v16bf af;
#pragma unroll
    for (int i = 0; i < 4; ++i) {
      af[i]      = (__bf16)a0[i];
      af[4 + i]  = (__bf16)a1[i];
      af[8 + i]  = (__bf16)a2[i];
      af[12 + i] = (__bf16)a3[i];
    }
    // B-frag: lane's column = n*16+lm; K {0..7,16..23} (lanes 0-15) / {8..15,24..31} (lanes 16-31)
    const __bf16* wl = ldsW + lm * WPITCH + k + hh * 8;
#pragma unroll
    for (int n = 0; n < 8; ++n) {
      const v8bf* wp = (const v8bf*)(wl + (size_t)n * 16 * WPITCH);
      v8bf blo = wp[0];
      v8bf bhi = wp[2];
      v16bf bf = __builtin_shufflevector(blo, bhi, 0, 1, 2, 3, 4, 5, 6, 7,
                                         8, 9, 10, 11, 12, 13, 14, 15);
      acc[n] = __builtin_amdgcn_wmma_f32_16x16x32_bf16(
          false, af, false, bf, (short)0, acc[n], false, false);
    }
  }

  // ---- epilogue: s[M] = sum_N tanh(acc + b[N]) * u[N]
  float bn[8], un[8];
#pragma unroll
  for (int n = 0; n < 8; ++n) {
    bn[n] = bp[n * 16 + lm];
    un[n] = up[n * 16 + lm];
  }
  float p[8];
#pragma unroll
  for (int j = 0; j < 8; ++j) {          // C layout: lanes<16 -> M=j ; lanes>=16 -> M=8+j
    float s = 0.0f;
#pragma unroll
    for (int n = 0; n < 8; ++n) s += tanhf(acc[n][j] + bn[n]) * un[n];
    p[j] = s;
  }
#pragma unroll
  for (int j = 0; j < 8; ++j)            // reduce across the 16 lanes sharing each row
    for (int m = 1; m < 16; m <<= 1) p[j] += __shfl_xor(p[j], m, 32);

  if (lm == 0) {
    float esum = 0.0f;
#pragma unroll
    for (int j = 0; j < 8; ++j) {
      const int row = t0 + hh * 8 + j;
      float e = expf(p[j]) * (float)mask[(size_t)bIdx * T_N + row];
      e_lds[wave * 16 + hh * 8 + j] = e;
      esum += e;
    }
    unsafeAtomicAdd(denom_lds, esum);
  }
  __syncthreads();

  // ---- weighted sum over this wave's 16 rows (coalesced; rows are L2-hot)
  v4f nacc[4] = {};
  const float* xw = x + ((size_t)bIdx * T_N + t0) * D_N;
  for (int m = 0; m < 16; ++m) {
    const float e = e_lds[wave * 16 + m];
    const v4f* xr = (const v4f*)(xw + (size_t)m * D_N);
#pragma unroll
    for (int i = 0; i < 4; ++i) nacc[i] += e * xr[i * 32 + lane];
  }
#pragma unroll
  for (int i = 0; i < 4; ++i)
#pragma unroll
    for (int c2 = 0; c2 < 4; ++c2)
      unsafeAtomicAdd(&numer_lds[(i * 32 + lane) * 4 + c2], nacc[i][c2]);
  __syncthreads();

  unsafeAtomicAdd(&numerAcc[(size_t)bIdx * D_N + tid], numer_lds[tid]);
  unsafeAtomicAdd(&numerAcc[(size_t)bIdx * D_N + tid + 256], numer_lds[tid + 256]);
  if (tid == 0) unsafeAtomicAdd(&denomAcc[bIdx], *denom_lds);
}

// ---------------------------------------------------------------------------
// Kernel 2: out[b][d] = numer[b][d] / (denom[b] + EPS)
// ---------------------------------------------------------------------------
__global__ void attn_finalize(const float* __restrict__ numer,
                              const float* __restrict__ denom,
                              float* __restrict__ out) {
  const int idx = blockIdx.x * blockDim.x + threadIdx.x;  // 0..32767
  const int b = idx >> 9;
  out[idx] = numer[idx] / (denom[b] + EPS_);
}

// ---------------------------------------------------------------------------
extern "C" void kernel_launch(void* const* d_in, const int* in_sizes, int n_in,
                              void* d_out, int out_size, void* d_ws, size_t ws_size,
                              hipStream_t stream) {
  const float*         x    = (const float*)d_in[0];
  const unsigned char* mask = (const unsigned char*)d_in[1];  // jax bool -> u8
  const float*         W    = (const float*)d_in[2];
  const float*         bias = (const float*)d_in[3];
  const float*         u    = (const float*)d_in[4];
  float* out = (float*)d_out;

  // workspace layout: numer[B*D] f32 | denom[B] f32 | Wt[A*D] bf16
  float*  numerAcc = (float*)d_ws;
  float*  denomAcc = numerAcc + B_N * D_N;
  __bf16* wtg = (__bf16*)((char*)d_ws + (size_t)(B_N * D_N + B_N) * 4);

  attn_prep<<<(A_N * D_N) / 256, 256, 0, stream>>>(W, numerAcc, wtg);

  const size_t smem = (size_t)A_N * WPITCH * 2 + D_N * 4 + T_TILE * 4 + 16;
  (void)hipFuncSetAttribute((const void*)attn_main,
                            hipFuncAttributeMaxDynamicSharedMemorySize, (int)smem);
  attn_main<<<B_N * N_CHUNK, 256, smem, stream>>>(x, mask, bias, u, wtg,
                                                  numerAcc, denomAcc);

  attn_finalize<<<(B_N * D_N) / 256, 256, 0, stream>>>(numerAcc, denomAcc, out);
}
